// AttentionLayer_62783831933054
// MI455X (gfx1250) — matrix operand order repaired and checked by reading, not compiled, not run on hardware
//
#include <hip/hip_runtime.h>
#include <hip/hip_bf16.h>

typedef __attribute__((ext_vector_type(16))) _Float16 v16h;
typedef __attribute__((ext_vector_type(8)))  _Float16 v8h;
typedef __attribute__((ext_vector_type(8)))  float    v8f;

#define B_SZ   2048
#define T_SZ   200
#define D_SZ   64
#define H0_SZ  80
#define H1_SZ  40
#define FIN    256
#define NEG_INF_F (-4294967295.0f)

#define W0P_ELEMS (5*8*32*16)   // 5 N-tiles x 8 K-steps x 32 lanes x 16 f16 = 20480
#define W1P_ELEMS (3*3*32*16)   // 3 N-tiles x 3 K-steps x 32 lanes x 16 f16 = 4608
#define WTS_ELEMS (W0P_ELEMS + W1P_ELEMS)

#define WAVES    4
#define NTILES_T 13             // ceil(200/16)

// ---------------------------------------------------------------------------
// Pre-pack W0 / W1 into f16 WMMA B-fragment layout:
// lane l of the fragment holds column n = nt*16 + (l&15),
// K values k = ks*32 + (l>>4)*8 + (i&7) + (i>>3)*16, i = 0..15.
// ---------------------------------------------------------------------------
__global__ void pack_weights_kernel(const float* __restrict__ W0,
                                    const float* __restrict__ W1,
                                    _Float16* __restrict__ w0p,
                                    _Float16* __restrict__ w1p)
{
    int tid = blockIdx.x * blockDim.x + threadIdx.x;
    int stride = gridDim.x * blockDim.x;

    for (int idx = tid; idx < W0P_ELEMS; idx += stride) {
        int i    = idx & 15;
        int lane = (idx >> 4) & 31;
        int ks   = (idx >> 9) & 7;
        int nt   = idx >> 12;
        int n = nt * 16 + (lane & 15);               // n < 80 always
        int kk = ks * 32 + ((lane >> 4) * 8) + (i & 7) + ((i >> 3) * 16);   // k < 256 always
        w0p[idx] = (_Float16)W0[kk * H0_SZ + n];
    }
    for (int idx = tid; idx < W1P_ELEMS; idx += stride) {
        int i    = idx & 15;
        int lane = (idx >> 4) & 31;
        int grp  = idx >> 9;          // nt*3 + ks, 0..8
        int ks   = grp % 3;
        int nt   = grp / 3;
        int n = nt * 16 + (lane & 15);
        int kk = ks * 32 + ((lane >> 4) * 8) + (i & 7) + ((i >> 3) * 16);
        float val = (n < H1_SZ && kk < H0_SZ) ? W1[kk * H1_SZ + n] : 0.0f;
        w1p[idx] = (_Float16)val;
    }
}

// ---------------------------------------------------------------------------
// Fused kernel: 1 block per batch element, 4 waves.
// B-fragments LDS-resident; A-fragments built in registers.
// ---------------------------------------------------------------------------
__global__ __launch_bounds__(128)
void attn_din_kernel(const float* __restrict__ q,
                     const float* __restrict__ k,
                     const float* __restrict__ v,
                     const _Float16* __restrict__ w0p,  // packed, w1p contiguous after
                     const float* __restrict__ b0,
                     const float* __restrict__ a0,
                     const float* __restrict__ b1,
                     const float* __restrict__ a1,
                     const float* __restrict__ Wo,
                     const float* __restrict__ bo,
                     const int*   __restrict__ mask,
                     float* __restrict__ out)
{
    __shared__ _Float16 s_wts[WTS_ELEMS];        // 50176 B: W0 then W1 fragments
    __shared__ _Float16 s_wave[WAVES][16 * 96];  // 3072 B/wave: k-tile (16x64) then h0 (16x96)
    __shared__ float s_logits[208];
    __shared__ float s_q[64];
    __shared__ float s_red[128];
    __shared__ float s_out[2][64];

    const int b    = blockIdx.x;
    const int tid  = threadIdx.x;
    const int wave = tid >> 5;
    const int lane = tid & 31;
    const int mrow = lane & 15;
    const int base = (lane >> 4) * 8;

    // ---- cooperative copy of packed weights into LDS (b128 copies) ----
    {
        const uint4* src = (const uint4*)w0p;    // 50176 B contiguous
        uint4* dst = (uint4*)s_wts;
        for (int i = tid; i < WTS_ELEMS / 8; i += 128) dst[i] = src[i];
    }
    if (tid < 64) s_q[tid] = q[b * 64 + tid];
    __syncthreads();

    // ---- q-side A fragments (cols 0..63), built once per block ----
    v16h aq0, aq1;
#pragma unroll
    for (int i = 0; i < 8; ++i) {
        aq0[i]     = (_Float16)s_q[base + i];
        aq0[i + 8] = (_Float16)s_q[base + 16 + i];
        aq1[i]     = (_Float16)s_q[32 + base + i];
        aq1[i + 8] = (_Float16)s_q[32 + base + 16 + i];
    }

    // hoist small per-lane scalars used by the logit stage
    const int n1i = 16 + mrow, n2i = 32 + mrow;
    const bool v2 = (n2i < H1_SZ);
    const float wo0 = Wo[mrow], b1_0 = b1[mrow], a1_0 = a1[mrow];
    const float wo1 = Wo[n1i],  b1_1 = b1[n1i],  a1_1 = a1[n1i];
    const float wo2 = v2 ? Wo[n2i] : 0.0f;
    const float b1_2 = v2 ? b1[n2i] : 0.0f;
    const float a1_2 = v2 ? a1[n2i] : 0.0f;
    const float bo0 = bo[0];

    _Float16* wbuf = &s_wave[wave][0];
    const _Float16* w0l = s_wts;
    const _Float16* w1l = s_wts + W0P_ELEMS;

    for (int tt = wave; tt < NTILES_T; tt += WAVES) {
        const int t0 = tt * 16;
        // stop LICM from hoisting the tile-invariant LDS B-loads into 320 regs
        asm volatile("" ::: "memory");

        // prefetch next k tile for this wave (global_prefetch_b8)
        {
            int tn = t0 + WAVES * 16;
            if (tn < T_SZ)
                __builtin_prefetch(&k[((size_t)b * T_SZ + tn) * D_SZ + (size_t)lane], 0, 0);
        }

        // ---- stage k tile as f16 (16 x 64) ----
        for (int idx = lane; idx < 16 * 64; idx += 32) {
            int r = idx >> 6, d = idx & 63;
            int t = t0 + r;
            float kv = (t < T_SZ) ? k[((size_t)b * T_SZ + t) * D_SZ + d] : 0.0f;
            wbuf[r * 64 + d] = (_Float16)kv;
        }

        // ---- k-side A fragments for this lane's row ----
        const _Float16* krow = wbuf + mrow * 64;
        v8h k0a = *(const v8h*)(krow + base);
        v8h k0b = *(const v8h*)(krow + base + 16);
        v8h k1a = *(const v8h*)(krow + 32 + base);
        v8h k1b = *(const v8h*)(krow + 32 + base + 16);
        v16h ak0 = __builtin_shufflevector(k0a, k0b, 0,1,2,3,4,5,6,7,8,9,10,11,12,13,14,15);
        v16h ak1 = __builtin_shufflevector(k1a, k1b, 0,1,2,3,4,5,6,7,8,9,10,11,12,13,14,15);

        // all 8 A fragments: [q | k | q-k | q*k] (packed f16 VALU for the last 4)
        v16h af[8];
        af[0] = aq0;        af[1] = aq1;
        af[2] = ak0;        af[3] = ak1;
        af[4] = aq0 - ak0;  af[5] = aq1 - ak1;
        af[6] = aq0 * ak0;  af[7] = aq1 * ak1;

        // ---- layer 0: [16x256] x [256x80], B fed from LDS ----
        v8f acc[5] = {};
#pragma unroll
        for (int ks = 0; ks < 8; ++ks) {
#pragma unroll
            for (int nt = 0; nt < 5; ++nt) {
                v16h bb = *(const v16h*)(w0l + (size_t)((nt * 8 + ks) * 32 + lane) * 16);
                acc[nt] = __builtin_amdgcn_wmma_f32_16x16x32_f16(
                    false, af[ks], false, bb, (short)0, acc[nt], false, false);
            }
        }

        // ---- bias + PReLU -> h0 (f16, rows padded to 96); overwrites k tile ----
#pragma unroll
        for (int nt = 0; nt < 5; ++nt) {
            int n = nt * 16 + mrow;
            float bias  = b0[n];
            float alpha = a0[n];
#pragma unroll
            for (int r = 0; r < 8; ++r) {
                int m = r + ((lane >> 4) * 8);
                float h = acc[nt][r] + bias;
                h = (h >= 0.0f) ? h : alpha * h;
                wbuf[m * 96 + n] = (_Float16)h;
            }
        }
        for (int idx = lane; idx < 16 * 16; idx += 32)
            wbuf[(idx >> 4) * 96 + 80 + (idx & 15)] = (_Float16)0.0f;

        // ---- layer 1: [16x96] x [96x48] (valid 80x40) ----
        v8f acc2[3] = {};
#pragma unroll
        for (int ks = 0; ks < 3; ++ks) {
            const _Float16* arow = wbuf + mrow * 96 + ks * 32 + base;
            v8h alo = *(const v8h*)(arow);
            v8h ahi = *(const v8h*)(arow + 16);
            v16h a = __builtin_shufflevector(alo, ahi, 0,1,2,3,4,5,6,7,8,9,10,11,12,13,14,15);
#pragma unroll
            for (int nt = 0; nt < 3; ++nt) {
                v16h bb = *(const v16h*)(w1l + (size_t)((nt * 3 + ks) * 32 + lane) * 16);
                acc2[nt] = __builtin_amdgcn_wmma_f32_16x16x32_f16(
                    false, a, false, bb, (short)0, acc2[nt], false, false);
            }
        }

        // ---- logits straight from accumulators via 16-lane xor reduction ----
#pragma unroll
        for (int r = 0; r < 8; ++r) {
            float h, p = 0.0f;
            h = acc2[0][r] + b1_0; h = (h >= 0.0f) ? h : a1_0 * h; p += h * wo0;
            h = acc2[1][r] + b1_1; h = (h >= 0.0f) ? h : a1_1 * h; p += h * wo1;
            h = acc2[2][r] + b1_2; h = (h >= 0.0f) ? h : a1_2 * h; if (v2) p += h * wo2;
            p += __shfl_xor(p, 1);
            p += __shfl_xor(p, 2);
            p += __shfl_xor(p, 4);
            p += __shfl_xor(p, 8);
            int t = t0 + r + ((lane >> 4) * 8);
            if (mrow == 0 && t < T_SZ) s_logits[t] = p + bo0;
        }
    }
    __syncthreads();

    // ---- masked softmax over T ----
    int ta = tid, tb = tid + 128;
    float lg0 = NEG_INF_F, lg1 = NEG_INF_F;
    if (ta < T_SZ) lg0 = (mask[b * T_SZ + ta] == 0) ? NEG_INF_F : s_logits[ta];
    if (tb < T_SZ) lg1 = (mask[b * T_SZ + tb] == 0) ? NEG_INF_F : s_logits[tb];

    s_red[tid] = fmaxf(lg0, lg1);
    __syncthreads();
    for (int s = 64; s > 0; s >>= 1) {
        if (tid < s) s_red[tid] = fmaxf(s_red[tid], s_red[tid + s]);
        __syncthreads();
    }
    float mx = s_red[0];
    __syncthreads();

    float e0 = (ta < T_SZ) ? __expf(lg0 - mx) : 0.0f;
    float e1 = (tb < T_SZ) ? __expf(lg1 - mx) : 0.0f;
    s_red[tid] = e0 + e1;
    __syncthreads();
    for (int s = 64; s > 0; s >>= 1) {
        if (tid < s) s_red[tid] += s_red[tid + s];
        __syncthreads();
    }
    float inv = 1.0f / s_red[0];
    __syncthreads();

    if (ta < T_SZ) s_logits[ta] = e0 * inv;
    if (tb < T_SZ) s_logits[tb] = e1 * inv;
    __syncthreads();

    // ---- out[b,d] = sum_t w_t * v[b,t,d] ----
    {
        int part = tid >> 6;   // 0 or 1
        int d    = tid & 63;
        float acc = 0.0f;
        int tbeg = part * 100, tend = tbeg + 100;
        for (int t = tbeg; t < tend; ++t)
            acc += s_logits[t] * v[((size_t)b * T_SZ + t) * D_SZ + d];
        s_out[part][d] = acc;
    }
    __syncthreads();
    if (tid < 64)
        out[b * 64 + tid] = s_out[0][tid] + s_out[1][tid];
}

// ---------------------------------------------------------------------------
extern "C" void kernel_launch(void* const* d_in, const int* in_sizes, int n_in,
                              void* d_out, int out_size, void* d_ws, size_t ws_size,
                              hipStream_t stream)
{
    const float* q    = (const float*)d_in[0];
    const float* k    = (const float*)d_in[1];
    const float* v    = (const float*)d_in[2];
    const float* W0   = (const float*)d_in[3];
    const float* b0   = (const float*)d_in[4];
    const float* a0   = (const float*)d_in[5];
    const float* W1   = (const float*)d_in[6];
    const float* b1   = (const float*)d_in[7];
    const float* a1   = (const float*)d_in[8];
    const float* Wo   = (const float*)d_in[9];
    const float* bo   = (const float*)d_in[10];
    const int*   mask = (const int*)d_in[11];

    _Float16* w0p = (_Float16*)d_ws;
    _Float16* w1p = w0p + W0P_ELEMS;   // contiguous after w0p

    pack_weights_kernel<<<40, 256, 0, stream>>>(W0, W1, w0p, w1p);
    attn_din_kernel<<<B_SZ, 128, 0, stream>>>(q, k, v, w0p, b0, a0,
                                              b1, a1, Wo, bo, mask,
                                              (float*)d_out);
}
